// NC2X_Model_19112604467789
// MI455X (gfx1250) — matrix operand order repaired and checked by reading, not compiled, not run on hardware
//
#include <hip/hip_runtime.h>
#include <hip/hip_bf16.h>

// ---------------- problem constants (from reference) ----------------
constexpr int N_NODES = 50000;
constexpr int N_EDGES = 200000;
constexpr int G_GRAPH = 64;
constexpr int F_IN    = 1536;
constexpr int H_DIM   = 1024;
constexpr int C_OUT   = 80;
constexpr int FUSE    = 1024;

typedef __attribute__((ext_vector_type(16))) __bf16 v16bf;
typedef __attribute__((ext_vector_type(8)))  float  v8f;
typedef __attribute__((ext_vector_type(4)))  unsigned int u32x4;
typedef __attribute__((ext_vector_type(8)))  int    i32x8;
typedef __attribute__((ext_vector_type(4)))  int    i32x4;

union FragBF { v16bf v; uint4 u[2]; };

// ---------------- elementwise / setup kernels ----------------
__global__ void fill_f32(float* __restrict__ p, float v, size_t n) {
    size_t i = (size_t)blockIdx.x * blockDim.x + threadIdx.x;
    if (i < n) p[i] = v;
}

__global__ void deg_add_k(const int* __restrict__ ei, float* __restrict__ deg, int E) {
    int e = blockIdx.x * blockDim.x + threadIdx.x;
    if (e < E) atomicAdd(&deg[ei[E + e]], 1.0f);   // dst = ei[1][e]
}

__global__ void rsqrt_k(const float* __restrict__ deg, float* __restrict__ dis, int n) {
    int i = blockIdx.x * blockDim.x + threadIdx.x;
    if (i < n) dis[i] = __frsqrt_rn(deg[i]);       // deg >= 1 (self loops)
}

__global__ void norm_k(const int* __restrict__ ei, const float* __restrict__ dis,
                       float* __restrict__ norm, int E, int N) {
    int i = blockIdx.x * blockDim.x + threadIdx.x;
    if (i >= E + N) return;
    int s, d;
    if (i < E) { s = ei[i]; d = ei[E + i]; } else { s = d = i - E; }
    norm[i] = dis[s] * dis[d];
}

__global__ void cvt_f32_bf16(const float* __restrict__ in, __bf16* __restrict__ out, size_t n) {
    size_t i = (size_t)blockIdx.x * blockDim.x + threadIdx.x;
    if (i < n) out[i] = (__bf16)in[i];
}

// fp32 [K,Nn] row-major  ->  bf16 [Nn,K] row-major (i.e. B^T, K-contiguous)
__global__ __launch_bounds__(256) void cvt_transpose_bf16(
    const float* __restrict__ W, __bf16* __restrict__ WT, int K, int Nn)
{
    __shared__ float t[32][33];
    int k0 = blockIdx.x * 32, n0 = blockIdx.y * 32;
    int tx = threadIdx.x & 31, ty = threadIdx.x >> 5;   // 8 row-groups
    for (int r = ty; r < 32; r += 8)
        t[r][tx] = W[(size_t)(k0 + r) * Nn + (n0 + tx)];
    __syncthreads();
    for (int r = ty; r < 32; r += 8)
        WT[(size_t)(n0 + r) * K + (k0 + tx)] = (__bf16)t[tx][r];
}

// ---------------- TDM descriptor helpers (D# per ISA 08_async_tensor §8) ----------------
__device__ __forceinline__ u32x4 tdm_g0(unsigned lds_addr, unsigned long long gaddr) {
    u32x4 g;
    g[0] = 1u;                                          // count=1, user mode
    g[1] = lds_addr;                                    // lds_addr [63:32]
    g[2] = (unsigned)(gaddr & 0xFFFFFFFFull);           // global_addr [95:64]
    g[3] = (unsigned)((gaddr >> 32) & 0x01FFFFFFull)    // global_addr [120:96]
         | (2u << 30);                                  // type=2 ("image")
    return g;
}

// 2D tile: dim0 = contiguous K, dim1 = rows; bf16 (2B); LDS pad every 16 DWORDs by 4 DWORDs
__device__ __forceinline__ i32x8 tdm_g1(unsigned tensor_d0, unsigned tensor_d1,
                                        unsigned tile_d0, unsigned tile_d1,
                                        unsigned long long d0_stride) {
    i32x8 g = {0, 0, 0, 0, 0, 0, 0, 0};
    g[0] = (int)((1u << 16)          // data_size = 2 bytes
               | (1u << 20)          // pad_enable
               | (3u << 22)          // pad_interval: 16 DWORDs
               | (3u << 25));        // pad_amount:   4 DWORDs (8 bf16)
    g[1] = (int)((tensor_d0 & 0xFFFFu) << 16);                          // tensor_dim0 [79:48]
    g[2] = (int)(((tensor_d0 >> 16) & 0xFFFFu)
               | ((tensor_d1 & 0xFFFFu) << 16));                        // tensor_dim1 [111:80]
    g[3] = (int)(((tensor_d1 >> 16) & 0xFFFFu)
               | ((tile_d0 & 0xFFFFu) << 16));                          // tile_dim0 [127:112]
    g[4] = (int)(tile_d1 & 0xFFFFu);                                    // tile_dim1 [143:128]
    g[5] = (int)(unsigned)(d0_stride & 0xFFFFFFFFull);                  // dim0_stride lo
    g[6] = (int)(unsigned)((d0_stride >> 32) & 0xFFFFull);              // dim0_stride hi
    return g;
}

// issue both A/B tile DMAs for k0 into LDS buffers at ldsA/ldsB (call from one wave)
__device__ __forceinline__ void tdm_issue_tiles(
    const __bf16* A, const __bf16* Bt, int bM, int bN, int M, int Nn, int K,
    int k0, unsigned ldsA, unsigned ldsB)
{
    const i32x4 z4 = {0, 0, 0, 0};
    const i32x8 z8 = {0, 0, 0, 0, 0, 0, 0, 0};
    unsigned long long gA = (unsigned long long)(uintptr_t)(A  + (size_t)bM * K + k0);
    unsigned long long gB = (unsigned long long)(uintptr_t)(Bt + (size_t)bN * K + k0);
    __builtin_amdgcn_tensor_load_to_lds(
        tdm_g0(ldsA, gA),
        tdm_g1((unsigned)(K - k0), (unsigned)(M - bM), 32u, 128u,
               (unsigned long long)K),
        z4, z4, z8, 0);
    __builtin_amdgcn_tensor_load_to_lds(
        tdm_g0(ldsB, gB),
        tdm_g1((unsigned)(K - k0), (unsigned)(Nn - bN), 32u, 128u,
               (unsigned long long)K),
        z4, z4, z8, 0);
}

// ---------------- bf16 WMMA GEMM: C[M,Nn] = A[M,K] @ Bt[Nn,K]^T ----------------
// Block tile 128x128, BK=32, 256 threads = 8 waves (4x2); each wave -> 32x64.
// Double-buffered LDS tiles staged by the Tensor Data Mover (wave 0 issues);
// DMA for step s+1 overlaps the 8 WMMAs of step s.
__global__ __launch_bounds__(256) void gemm_bf16_wmma(
    const __bf16* __restrict__ A,   // [M,K] row-major
    const __bf16* __restrict__ Bt,  // [Nn,K] row-major (B transposed)
    float* __restrict__ C, int M, int K, int Nn)
{
    __shared__ __bf16 As[2][128][40];  // 80B rows, 16B-aligned fragment chunks
    __shared__ __bf16 Bs[2][128][40];

    const int tid   = threadIdx.x;
    const int lane  = tid & 31;
    const int wave  = tid >> 5;     // 0..7
    const int waveM = wave >> 1;    // 0..3  (rows of 32)
    const int waveN = wave & 1;     // 0..1  (cols of 64)
    const int bM = blockIdx.x * 128;
    const int bN = blockIdx.y * 128;

    const int lrow = lane & 15;     // A row / B,C column within 16
    const int lhi  = lane >> 4;     // half-wave select
    const int kbA  = lhi * 8;       // A frag: K kb..kb+7 then kb+16..kb+23
    const int kbB  = lhi * 8;       // B frag: K kb..kb+7 then kb+16..kb+23

    const unsigned ldsA0 = (unsigned)(uintptr_t)&As[0][0][0];
    const unsigned ldsA1 = (unsigned)(uintptr_t)&As[1][0][0];
    const unsigned ldsB0 = (unsigned)(uintptr_t)&Bs[0][0][0];
    const unsigned ldsB1 = (unsigned)(uintptr_t)&Bs[1][0][0];

    v8f acc[2][4] = {};

    const int nsteps = K / 32;

    // prologue: fill buffer 0
    if (wave == 0) {
        tdm_issue_tiles(A, Bt, bM, bN, M, Nn, K, 0, ldsA0, ldsB0);
        __builtin_amdgcn_s_wait_tensorcnt(0);
    }
    __syncthreads();

    for (int s = 0; s < nsteps; ++s) {
        const int cur = s & 1;

        // ---- prefetch next tiles into the other buffer (overlaps WMMAs) ----
        if (wave == 0 && (s + 1) < nsteps) {
            tdm_issue_tiles(A, Bt, bM, bN, M, Nn, K, (s + 1) * 32,
                            cur ? ldsA0 : ldsA1, cur ? ldsB0 : ldsB1);
        }

        // ---- fragments: pure b128 LDS traffic ----
        FragBF a[2], b[4];
#pragma unroll
        for (int mi = 0; mi < 2; ++mi) {
            int r = waveM * 32 + mi * 16 + lrow;
            a[mi].u[0] = *(const uint4*)&As[cur][r][kbA];        // K kb..kb+7
            a[mi].u[1] = *(const uint4*)&As[cur][r][kbA + 16];   // K kb+16..kb+23
        }
#pragma unroll
        for (int ni = 0; ni < 4; ++ni) {
            int c = waveN * 64 + ni * 16 + lrow;
            b[ni].u[0] = *(const uint4*)&Bs[cur][c][kbB];        // K kb..kb+7
            b[ni].u[1] = *(const uint4*)&Bs[cur][c][kbB + 16];   // K kb+16..kb+23
        }
        // ---- 8 WMMAs per wave per k-step, fp32 accumulate ----
#pragma unroll
        for (int mi = 0; mi < 2; ++mi)
#pragma unroll
            for (int ni = 0; ni < 4; ++ni)
                acc[mi][ni] = __builtin_amdgcn_wmma_f32_16x16x32_bf16(
                    false, a[mi].v, false, b[ni].v,
                    (short)0, acc[mi][ni], false, false);

        // ---- next buffer must be resident before anyone reads it ----
        if (wave == 0) __builtin_amdgcn_s_wait_tensorcnt(0);
        __syncthreads();
    }

    // ---- epilogue: ISA C/D layout (VGPR i -> row i + 8*half) ----
#pragma unroll
    for (int mi = 0; mi < 2; ++mi) {
#pragma unroll
        for (int ni = 0; ni < 4; ++ni) {
            int col = bN + waveN * 64 + ni * 16 + lrow;
#pragma unroll
            for (int i = 0; i < 8; ++i) {
                int row = bM + waveM * 32 + mi * 16 + lhi * 8 + i;
                if (row < M) C[(size_t)row * Nn + col] = acc[mi][ni][i];
            }
        }
    }
}

// ---------------- graph scatter: out[dst] += h[src] * norm (incl. self loops) ----------------
__global__ __launch_bounds__(256) void scatter_k(
    const int* __restrict__ ei, const float* __restrict__ norm,
    const float* __restrict__ h, float* __restrict__ out, int E, int N)
{
    int e = blockIdx.x;
    int s, d;
    if (e < E) { s = ei[e]; d = ei[E + e]; } else { s = d = e - E; }
    float w = norm[e];
    const float4* hs = (const float4*)(h + (size_t)s * H_DIM);
    float* od = out + (size_t)d * H_DIM;
    for (int c = threadIdx.x; c < H_DIM / 4; c += blockDim.x) {
        float4 v = hs[c];
        atomicAdd(&od[c * 4 + 0], v.x * w);
        atomicAdd(&od[c * 4 + 1], v.y * w);
        atomicAdd(&od[c * 4 + 2], v.z * w);
        atomicAdd(&od[c * 4 + 3], v.w * w);
    }
}

__global__ void bias_relu_k(float* __restrict__ p, const float* __restrict__ b, size_t n, int H) {
    size_t i = (size_t)blockIdx.x * blockDim.x + threadIdx.x;
    if (i < n) p[i] = fmaxf(p[i] + b[i % H], 0.0f);
}

// ---------------- mean pool per graph ----------------
__global__ __launch_bounds__(256) void pool_k(
    const float* __restrict__ h, const int* __restrict__ batch,
    float* __restrict__ pooled, float* __restrict__ counts)
{
    int nidx = blockIdx.x;
    int g = batch[nidx];
    const float* hr = h + (size_t)nidx * H_DIM;
    float* pr = pooled + (size_t)g * H_DIM;
    for (int c = threadIdx.x; c < H_DIM; c += blockDim.x)
        atomicAdd(&pr[c], hr[c]);
    if (threadIdx.x == 0) atomicAdd(&counts[g], 1.0f);
}

__global__ void pool_div_k(float* __restrict__ pooled, const float* __restrict__ counts) {
    int g = blockIdx.x;
    float inv = 1.0f / fmaxf(counts[g], 1.0f);
    for (int c = threadIdx.x; c < H_DIM; c += blockDim.x)
        pooled[(size_t)g * H_DIM + c] *= inv;
}

// ---------------- small fused MLP head (64 rows; VALU is fine) ----------------
__global__ __launch_bounds__(256) void mlp1_k(
    const float* __restrict__ gx, const float* __restrict__ pooled,
    const float* __restrict__ fw1, const float* __restrict__ fb1,
    float* __restrict__ hid)
{
    __shared__ float comb[F_IN + H_DIM];   // 2560 floats
    int g = blockIdx.x;
    for (int i = threadIdx.x; i < F_IN; i += blockDim.x)  comb[i] = gx[(size_t)g * F_IN + i];
    for (int i = threadIdx.x; i < H_DIM; i += blockDim.x) comb[F_IN + i] = pooled[(size_t)g * H_DIM + i];
    __syncthreads();
    for (int j = threadIdx.x; j < FUSE; j += blockDim.x) {
        float acc = fb1[j];
        for (int k = 0; k < F_IN + H_DIM; ++k)
            acc = fmaf(comb[k], fw1[(size_t)k * FUSE + j], acc);
        hid[(size_t)g * FUSE + j] = fmaxf(acc, 0.0f);
    }
}

__global__ __launch_bounds__(128) void mlp2_k(
    const float* __restrict__ hid, const float* __restrict__ fw2,
    const float* __restrict__ fb2, float* __restrict__ out)
{
    __shared__ float hrow[FUSE];
    int g = blockIdx.x;
    for (int i = threadIdx.x; i < FUSE; i += blockDim.x) hrow[i] = hid[(size_t)g * FUSE + i];
    __syncthreads();
    for (int j = threadIdx.x; j < C_OUT; j += blockDim.x) {
        float acc = fb2[j];
        for (int k = 0; k < FUSE; ++k)
            acc = fmaf(hrow[k], fw2[(size_t)k * C_OUT + j], acc);
        out[(size_t)g * C_OUT + j] = acc;
    }
}

// ---------------- launch ----------------
extern "C" void kernel_launch(void* const* d_in, const int* in_sizes, int n_in,
                              void* d_out, int out_size, void* d_ws, size_t ws_size,
                              hipStream_t stream) {
    const float* gx  = (const float*)d_in[0];
    const float* x   = (const float*)d_in[1];
    const int*   ei  = (const int*)  d_in[2];
    const int*   bat = (const int*)  d_in[3];
    const float* W1  = (const float*)d_in[4];
    const float* b1  = (const float*)d_in[5];
    const float* W2  = (const float*)d_in[6];
    const float* b2  = (const float*)d_in[7];
    const float* fw1 = (const float*)d_in[8];
    const float* fb1 = (const float*)d_in[9];
    const float* fw2 = (const float*)d_in[10];
    const float* fb2 = (const float*)d_in[11];
    float* out = (float*)d_out;

    // workspace carve-out
    char* w = (char*)d_ws;
    auto alloc = [&](size_t bytes) -> void* {
        void* p = (void*)w; w += (bytes + 255) & ~(size_t)255; return p;
    };
    float*  deg    = (float*) alloc((size_t)N_NODES * 4);
    float*  dis    = (float*) alloc((size_t)N_NODES * 4);
    float*  nrm    = (float*) alloc((size_t)(N_EDGES + N_NODES) * 4);
    float*  counts = (float*) alloc((size_t)G_GRAPH * 4);
    float*  pooled = (float*) alloc((size_t)G_GRAPH * H_DIM * 4);
    float*  hid    = (float*) alloc((size_t)G_GRAPH * FUSE * 4);
    __bf16* w1t    = (__bf16*)alloc((size_t)F_IN * H_DIM * 2);    // W1^T [H, F]
    __bf16* w2t    = (__bf16*)alloc((size_t)H_DIM * H_DIM * 2);   // W2^T [H, H]
    __bf16* xb     = (__bf16*)alloc((size_t)N_NODES * F_IN * 2);  // reused as h_bf16
    float*  bufA   = (float*) alloc((size_t)N_NODES * H_DIM * 4); // GEMM output
    float*  bufB   = (float*) alloc((size_t)N_NODES * H_DIM * 4); // scatter output

    const int EPN = N_EDGES + N_NODES;
    const size_t NH = (size_t)N_NODES * H_DIM;
    const size_t NF = (size_t)N_NODES * F_IN;
    auto gsz = [](size_t n, int b) { return (unsigned)((n + b - 1) / b); };

    // 1) degrees (init to 1 for self-loops), rsqrt, per-edge norm
    fill_f32<<<gsz(N_NODES,256),256,0,stream>>>(deg, 1.0f, N_NODES);
    deg_add_k<<<gsz(N_EDGES,256),256,0,stream>>>(ei, deg, N_EDGES);
    rsqrt_k<<<gsz(N_NODES,256),256,0,stream>>>(deg, dis, N_NODES);
    norm_k<<<gsz(EPN,256),256,0,stream>>>(ei, dis, nrm, N_EDGES, N_NODES);

    // 2) bf16 cast of x, bf16 transposed weights
    cvt_f32_bf16<<<gsz(NF,256),256,0,stream>>>(x, xb, NF);
    cvt_transpose_bf16<<<dim3(F_IN/32, H_DIM/32),256,0,stream>>>(W1, w1t, F_IN, H_DIM);
    cvt_transpose_bf16<<<dim3(H_DIM/32, H_DIM/32),256,0,stream>>>(W2, w2t, H_DIM, H_DIM);

    dim3 gemmGrid(gsz(N_NODES, 128), H_DIM / 128);

    // 3) layer 1: GEMM -> scatter -> bias+relu
    gemm_bf16_wmma<<<gemmGrid,256,0,stream>>>(xb, w1t, bufA, N_NODES, F_IN, H_DIM);
    fill_f32<<<gsz(NH,256),256,0,stream>>>(bufB, 0.0f, NH);
    scatter_k<<<EPN,256,0,stream>>>(ei, nrm, bufA, bufB, N_EDGES, N_NODES);
    bias_relu_k<<<gsz(NH,256),256,0,stream>>>(bufB, b1, NH, H_DIM);

    // 4) layer 2
    cvt_f32_bf16<<<gsz(NH,256),256,0,stream>>>(bufB, xb, NH);  // reuse xb as h_bf16
    gemm_bf16_wmma<<<gemmGrid,256,0,stream>>>(xb, w2t, bufA, N_NODES, H_DIM, H_DIM);
    fill_f32<<<gsz(NH,256),256,0,stream>>>(bufB, 0.0f, NH);
    scatter_k<<<EPN,256,0,stream>>>(ei, nrm, bufA, bufB, N_EDGES, N_NODES);
    bias_relu_k<<<gsz(NH,256),256,0,stream>>>(bufB, b2, NH, H_DIM);

    // 5) mean pool
    fill_f32<<<gsz((size_t)G_GRAPH*H_DIM,256),256,0,stream>>>(pooled, 0.0f, (size_t)G_GRAPH*H_DIM);
    fill_f32<<<1,256,0,stream>>>(counts, 0.0f, G_GRAPH);
    pool_k<<<N_NODES,256,0,stream>>>(bufB, bat, pooled, counts);
    pool_div_k<<<G_GRAPH,256,0,stream>>>(pooled, counts);

    // 6) fused MLP head
    mlp1_k<<<G_GRAPH,256,0,stream>>>(gx, pooled, fw1, fb1, hid);
    mlp2_k<<<G_GRAPH,128,0,stream>>>(hid, fw2, fb2, out);
}
